// PretrainedCROMA_3049426780847
// MI455X (gfx1250) — compile-verified, matrix operand order checked by reading, not run on hardware
//
#include <hip/hip_runtime.h>
#include <hip/hip_bf16.h>
#include <math.h>

// ---------------------------------------------------------------------------
// CROMA forward on gfx1250: all GEMMs via v_wmma_f32_16x16x32_bf16 (wave32).
// Staging: branchless float4 global loads (row-clamped; discarded rows absorb
// garbage) -> truncation-packed bf16 pairs -> LDS b64/b32 stores;
// fragments: two ds_load_b128 per 16x32/32x16 operand (ISA 05_wmma layout).
// ---------------------------------------------------------------------------

typedef __bf16 bf16_t;
typedef __attribute__((ext_vector_type(16))) __bf16 v16bf;
typedef __attribute__((ext_vector_type(8)))  float  v8f;

#define NTOK   225          // patches per image
#define DIMM   768
#define NHEAD  16
#define HDIM   48
#define SLD    256          // padded row stride of the score matrix (zeros in pad)
#define LDSS   40           // LDS row stride (bf16), keeps 16B alignment

// Truncation to bf16 (cheap: 1-2 VALU per pair, often a single v_perm_b32).
__device__ __forceinline__ unsigned pack2bf(float lo, float hi) {
    unsigned ul = __builtin_bit_cast(unsigned, lo);
    unsigned uh = __builtin_bit_cast(unsigned, hi);
    return (uh & 0xffff0000u) | (ul >> 16);
}

struct Frag32 { uint4 lo, hi; };
__device__ __forceinline__ v16bf load_frag(const bf16_t* p) {
    Frag32 f;
    f.lo = *reinterpret_cast<const uint4*>(p);        // K = k0 .. k0+7
    f.hi = *reinterpret_cast<const uint4*>(p + 16);   // K = k0+16 .. k0+23
    return __builtin_bit_cast(v16bf, f);
}

__device__ __forceinline__ float gelu_erf(float x) {
    return 0.5f * x * (1.0f + erff(x * 0.70710678118654752f));
}

// ---------------------------------------------------------------------------
// Generic GEMM: C[M,N] = A[M,K] @ W[K,N] (+bias) (+GELU) (+residual)
// Block tile 64x128, 8 waves (2Mx4N), wave tile 32x32 = 2x2 WMMA 16x16x32.
// Requires: K % 32 == 0, N % 128 == 0 (true for every call below).
// flags: 1 = add bias, 2 = gelu, 4 = add residual
// ---------------------------------------------------------------------------
__global__ __launch_bounds__(256) void gemm_wmma(
    const float* __restrict__ A, int lda,
    const float* __restrict__ W, int ldw,
    const float* __restrict__ bias,
    const float* __restrict__ res, int ldres,
    float* __restrict__ C, int ldc,
    int M, int N, int K, int flags)
{
    __shared__ bf16_t sA[64 * LDSS];
    __shared__ bf16_t sB[128 * LDSS];

    const int m0 = blockIdx.y * 64;
    const int n0 = blockIdx.x * 128;
    const int t = threadIdx.x;
    const int lane = t & 31, wave = t >> 5;
    const int waveM = wave & 1, waveN = wave >> 1;

    v8f acc[2][2] = {};

    for (int kt = 0; kt < K; kt += 32) {
        // A tile 64x32: 512 float4, 2 per thread. Row index clamped: garbage
        // rows only feed accumulator rows the store-guard discards.
        #pragma unroll
        for (int i = 0; i < 2; ++i) {
            int li = i * 256 + t;
            int r = li >> 3, c4 = (li & 7) << 2;
            int gm = m0 + r; gm = (gm < M) ? gm : (M - 1);
            float4 v = *reinterpret_cast<const float4*>(&A[(size_t)gm * lda + kt + c4]);
            uint2 p; p.x = pack2bf(v.x, v.y); p.y = pack2bf(v.z, v.w);
            *reinterpret_cast<uint2*>(&sA[r * LDSS + c4]) = p;
        }
        // W tile 32x128 -> LDS [n][k]: k-pairs, branchless (N%128==0, K%32==0)
        #pragma unroll
        for (int i = 0; i < 2; ++i) {
            int li = i * 256 + t;
            int kp = li >> 5;              // 0..15 (pair of k rows)
            int n4 = (li & 31) << 2;       // 0..124
            const float* w0 = &W[(size_t)(kt + 2 * kp) * ldw + n0 + n4];
            float4 a0 = *reinterpret_cast<const float4*>(w0);
            float4 a1 = *reinterpret_cast<const float4*>(w0 + ldw);
            *reinterpret_cast<unsigned*>(&sB[(n4 + 0) * LDSS + 2 * kp]) = pack2bf(a0.x, a1.x);
            *reinterpret_cast<unsigned*>(&sB[(n4 + 1) * LDSS + 2 * kp]) = pack2bf(a0.y, a1.y);
            *reinterpret_cast<unsigned*>(&sB[(n4 + 2) * LDSS + 2 * kp]) = pack2bf(a0.z, a1.z);
            *reinterpret_cast<unsigned*>(&sB[(n4 + 3) * LDSS + 2 * kp]) = pack2bf(a0.w, a1.w);
        }
        __syncthreads();

        const int k0 = (lane >> 4) << 3;   // ISA 16-bit operand layout
        v16bf aF[2], bF[2];
        #pragma unroll
        for (int mt = 0; mt < 2; ++mt)
            aF[mt] = load_frag(&sA[(waveM * 32 + mt * 16 + (lane & 15)) * LDSS + k0]);
        #pragma unroll
        for (int nt = 0; nt < 2; ++nt)
            bF[nt] = load_frag(&sB[(waveN * 32 + nt * 16 + (lane & 15)) * LDSS + k0]);
        #pragma unroll
        for (int mt = 0; mt < 2; ++mt)
            #pragma unroll
            for (int nt = 0; nt < 2; ++nt)
                acc[mt][nt] = __builtin_amdgcn_wmma_f32_16x16x32_bf16(
                    false, aF[mt], false, bF[nt], (short)0, acc[mt][nt], false, false);
        __syncthreads();
    }

    #pragma unroll
    for (int mt = 0; mt < 2; ++mt)
        #pragma unroll
        for (int nt = 0; nt < 2; ++nt) {
            int n = n0 + waveN * 32 + nt * 16 + (lane & 15);
            float bv = (flags & 1) ? bias[n] : 0.0f;
            int rowBase = m0 + waveM * 32 + mt * 16 + ((lane >> 4) << 3);
            #pragma unroll
            for (int r = 0; r < 8; ++r) {
                int m = rowBase + r;
                if (m >= M) continue;
                float v = acc[mt][nt][r] + bv;
                if (flags & 2) v = gelu_erf(v);
                if (flags & 4) v += res[(size_t)m * ldres + n];
                C[(size_t)m * ldc + n] = v;
            }
        }
}

// ---------------------------------------------------------------------------
// Attention scores: S[bh,i,j] = scale * q_i . k_j + bias[h,i,j]
// q,k packed in qkv[B*225, 2304]; HD=48 padded to K=64 (c4<dlim float4 guard,
// folds away for kt=0 after full unroll). Token rows/cols clamped (discarded).
// grid: (2 n-tiles, 4 m-tiles, B*H), block 256.
// ---------------------------------------------------------------------------
__global__ __launch_bounds__(256) void attn_scores_wmma(
    const float* __restrict__ qkv, const float* __restrict__ bias,
    float* __restrict__ S, float scale)
{
    __shared__ bf16_t sA[64 * LDSS];
    __shared__ bf16_t sB[128 * LDSS];

    const int bh = blockIdx.z, b = bh >> 4, h = bh & 15;
    const int m0 = blockIdx.y * 64;
    const int n0 = blockIdx.x * 128;
    const int t = threadIdx.x;
    const int lane = t & 31, wave = t >> 5;
    const int waveM = wave & 1, waveN = wave >> 1;
    const int qoff = h * HDIM, koff = DIMM + h * HDIM;

    v8f acc[2][2] = {};

    #pragma unroll
    for (int kt = 0; kt < 64; kt += 32) {
        const int dlim = HDIM - kt;        // 48 then 16, float4-aligned
        // q tile 64x32 (d contiguous)
        #pragma unroll
        for (int i = 0; i < 2; ++i) {
            int li = i * 256 + t;
            int r = li >> 3, c4 = (li & 7) << 2;
            int gm = m0 + r; gm = (gm < NTOK) ? gm : (NTOK - 1);
            float4 v = make_float4(0.f, 0.f, 0.f, 0.f);
            if (c4 < dlim)                 // K-dim pad must be exact zeros
                v = *reinterpret_cast<const float4*>(
                        &qkv[(size_t)(b * NTOK + gm) * 2304 + qoff + kt + c4]);
            uint2 p; p.x = pack2bf(v.x, v.y); p.y = pack2bf(v.z, v.w);
            *reinterpret_cast<uint2*>(&sA[r * LDSS + c4]) = p;
        }
        // k tile 128x32 (d contiguous; LDS also d contiguous -> no transpose)
        #pragma unroll
        for (int i = 0; i < 4; ++i) {
            int li = i * 256 + t;
            int r = li >> 3, c4 = (li & 7) << 2;
            int gj = n0 + r; gj = (gj < NTOK) ? gj : (NTOK - 1);
            float4 v = make_float4(0.f, 0.f, 0.f, 0.f);
            if (c4 < dlim)
                v = *reinterpret_cast<const float4*>(
                        &qkv[(size_t)(b * NTOK + gj) * 2304 + koff + kt + c4]);
            uint2 p; p.x = pack2bf(v.x, v.y); p.y = pack2bf(v.z, v.w);
            *reinterpret_cast<uint2*>(&sB[r * LDSS + c4]) = p;
        }
        __syncthreads();

        const int k0 = (lane >> 4) << 3;
        v16bf aF[2], bF[2];
        #pragma unroll
        for (int mt = 0; mt < 2; ++mt)
            aF[mt] = load_frag(&sA[(waveM * 32 + mt * 16 + (lane & 15)) * LDSS + k0]);
        #pragma unroll
        for (int nt = 0; nt < 2; ++nt)
            bF[nt] = load_frag(&sB[(waveN * 32 + nt * 16 + (lane & 15)) * LDSS + k0]);
        #pragma unroll
        for (int mt = 0; mt < 2; ++mt)
            #pragma unroll
            for (int nt = 0; nt < 2; ++nt)
                acc[mt][nt] = __builtin_amdgcn_wmma_f32_16x16x32_bf16(
                    false, aF[mt], false, bF[nt], (short)0, acc[mt][nt], false, false);
        __syncthreads();
    }

    #pragma unroll
    for (int mt = 0; mt < 2; ++mt)
        #pragma unroll
        for (int nt = 0; nt < 2; ++nt) {
            int j = n0 + waveN * 32 + nt * 16 + (lane & 15);
            if (j >= NTOK) continue;
            int rowBase = m0 + waveM * 32 + mt * 16 + ((lane >> 4) << 3);
            #pragma unroll
            for (int r = 0; r < 8; ++r) {
                int i = rowBase + r;
                if (i >= NTOK) continue;
                float v = acc[mt][nt][r] * scale
                        + bias[((size_t)h * NTOK + i) * NTOK + j];
                S[((size_t)bh * NTOK + i) * SLD + j] = v;
            }
        }
}

// ---------------------------------------------------------------------------
// Row softmax over 225 entries (ld = 256); writes zeros into the padding so
// the AV kernel can stage its A-tile branchless.
// ---------------------------------------------------------------------------
__global__ __launch_bounds__(256) void softmax_rows(float* __restrict__ S, int nrows)
{
    const int wave = threadIdx.x >> 5, lane = threadIdx.x & 31;
    const int row = blockIdx.x * 8 + wave;
    if (row >= nrows) return;
    float* p = S + (size_t)row * SLD;
    float vals[8];
    float mx = -1e30f;
    #pragma unroll
    for (int tt = 0; tt < 8; ++tt) {
        int j = lane + tt * 32;
        float v = (j < NTOK) ? p[j] : -1e30f;
        vals[tt] = v; mx = fmaxf(mx, v);
    }
    for (int o = 16; o > 0; o >>= 1) mx = fmaxf(mx, __shfl_xor(mx, o, 32));
    float sum = 0.0f;
    #pragma unroll
    for (int tt = 0; tt < 8; ++tt) {
        int j = lane + tt * 32;
        if (j < NTOK) { vals[tt] = expf(vals[tt] - mx); sum += vals[tt]; }
    }
    for (int o = 16; o > 0; o >>= 1) sum += __shfl_xor(sum, o, 32);
    float inv = 1.0f / sum;
    #pragma unroll
    for (int tt = 0; tt < 8; ++tt) {
        int j = lane + tt * 32;
        p[j] = (j < NTOK) ? vals[tt] * inv : 0.0f;   // zero the pad (j<256)
    }
}

// ---------------------------------------------------------------------------
// AV: O[b,i,h*48+n] = sum_j P[bh,i,j] * v[b,j,h,n]. K = 225 (pad zeroed).
// grid: (1, 4 m-tiles, B*H), block 128 (4 waves, 2x2), tile 64x64 (n<48 kept).
// P staging fully branchless (row clamp + zeroed pad); V keeps k-guards only.
// ---------------------------------------------------------------------------
__global__ __launch_bounds__(128) void attn_av_wmma(
    const float* __restrict__ S, const float* __restrict__ qkv,
    float* __restrict__ O)
{
    __shared__ bf16_t sA[64 * LDSS];
    __shared__ bf16_t sB[64 * LDSS];

    const int bh = blockIdx.z, b = bh >> 4, h = bh & 15;
    const int m0 = blockIdx.y * 64;
    const int t = threadIdx.x;
    const int lane = t & 31, wave = t >> 5;
    const int waveM = wave & 1, waveN = wave >> 1;
    const int voff = 2 * DIMM + h * HDIM;

    v8f acc[2][2] = {};

    for (int kt = 0; kt < NTOK; kt += 32) {
        // P tile 64x32: pad columns are exact zeros, rows clamped
        #pragma unroll
        for (int i = 0; i < 4; ++i) {
            int li = i * 128 + t;
            int r = li >> 3, c4 = (li & 7) << 2;
            int gm = m0 + r; gm = (gm < NTOK) ? gm : (NTOK - 1);
            float4 v = *reinterpret_cast<const float4*>(
                    &S[((size_t)bh * NTOK + gm) * SLD + kt + c4]);
            uint2 p; p.x = pack2bf(v.x, v.y); p.y = pack2bf(v.z, v.w);
            *reinterpret_cast<uint2*>(&sA[r * LDSS + c4]) = p;
        }
        // V tile 32x64 -> LDS [n][k] via k-pairs; K-dim guard required
        #pragma unroll
        for (int i = 0; i < 2; ++i) {
            int li = i * 128 + t;           // 0..255
            int kp = li >> 4;               // 0..15
            int n4 = (li & 15) << 2;        // 0..60
            int gk0 = kt + 2 * kp, gk1 = gk0 + 1;
            float4 a0 = make_float4(0.f, 0.f, 0.f, 0.f);
            float4 a1 = a0;
            if (gk0 < NTOK)
                a0 = *reinterpret_cast<const float4*>(
                        &qkv[(size_t)(b * NTOK + gk0) * 2304 + voff + n4]);
            if (gk1 < NTOK)
                a1 = *reinterpret_cast<const float4*>(
                        &qkv[(size_t)(b * NTOK + gk1) * 2304 + voff + n4]);
            *reinterpret_cast<unsigned*>(&sB[(n4 + 0) * LDSS + 2 * kp]) = pack2bf(a0.x, a1.x);
            *reinterpret_cast<unsigned*>(&sB[(n4 + 1) * LDSS + 2 * kp]) = pack2bf(a0.y, a1.y);
            *reinterpret_cast<unsigned*>(&sB[(n4 + 2) * LDSS + 2 * kp]) = pack2bf(a0.z, a1.z);
            *reinterpret_cast<unsigned*>(&sB[(n4 + 3) * LDSS + 2 * kp]) = pack2bf(a0.w, a1.w);
        }
        __syncthreads();

        const int k0 = (lane >> 4) << 3;
        v16bf aF[2], bF[2];
        #pragma unroll
        for (int mt = 0; mt < 2; ++mt)
            aF[mt] = load_frag(&sA[(waveM * 32 + mt * 16 + (lane & 15)) * LDSS + k0]);
        #pragma unroll
        for (int nt = 0; nt < 2; ++nt)
            bF[nt] = load_frag(&sB[(waveN * 32 + nt * 16 + (lane & 15)) * LDSS + k0]);
        #pragma unroll
        for (int mt = 0; mt < 2; ++mt)
            #pragma unroll
            for (int nt = 0; nt < 2; ++nt)
                acc[mt][nt] = __builtin_amdgcn_wmma_f32_16x16x32_bf16(
                    false, aF[mt], false, bF[nt], (short)0, acc[mt][nt], false, false);
        __syncthreads();
    }

    #pragma unroll
    for (int mt = 0; mt < 2; ++mt)
        #pragma unroll
        for (int nt = 0; nt < 2; ++nt) {
            int n = waveN * 32 + nt * 16 + (lane & 15);
            if (n >= HDIM) continue;
            int rowBase = m0 + waveM * 32 + mt * 16 + ((lane >> 4) << 3);
            #pragma unroll
            for (int r = 0; r < 8; ++r) {
                int m = rowBase + r;
                if (m >= NTOK) continue;
                O[(size_t)(b * NTOK + m) * DIMM + h * HDIM + n] = acc[mt][nt][r];
            }
        }
}

// ---------------------------------------------------------------------------
// LayerNorm over last dim 768. One wave per row.
// ---------------------------------------------------------------------------
__global__ __launch_bounds__(256) void layernorm_rows(
    const float* __restrict__ X, const float* __restrict__ g,
    const float* __restrict__ bb, float* __restrict__ Y, int rows)
{
    const int wave = threadIdx.x >> 5, lane = threadIdx.x & 31;
    const int row = blockIdx.x * 8 + wave;
    if (row >= rows) return;
    const float* x = X + (size_t)row * DIMM;
    float* y = Y + (size_t)row * DIMM;
    float s = 0.0f;
    for (int j = lane; j < DIMM; j += 32) s += x[j];
    for (int o = 16; o > 0; o >>= 1) s += __shfl_xor(s, o, 32);
    float mean = s * (1.0f / DIMM);
    float vs = 0.0f;
    for (int j = lane; j < DIMM; j += 32) { float d = x[j] - mean; vs += d * d; }
    for (int o = 16; o > 0; o >>= 1) vs += __shfl_xor(vs, o, 32);
    float inv = rsqrtf(vs * (1.0f / DIMM) + 1e-5f);
    for (int j = lane; j < DIMM; j += 32)
        y[j] = (x[j] - mean) * inv * g[j] + bb[j];
}

__global__ __launch_bounds__(256) void mean_pool_k(
    const float* __restrict__ X, float* __restrict__ Y, int nb)
{
    int t = blockIdx.x * blockDim.x + threadIdx.x;
    if (t >= nb * DIMM) return;
    int b = t / DIMM, d = t % DIMM;
    float s = 0.0f;
    for (int i = 0; i < NTOK; ++i) s += X[(size_t)(b * NTOK + i) * DIMM + d];
    Y[t] = s * (1.0f / NTOK);
}

__global__ __launch_bounds__(256) void patchify_k(
    const float* __restrict__ img, float* __restrict__ out, int C, int total)
{
    int t = blockIdx.x * blockDim.x + threadIdx.x;
    if (t >= total) return;
    int inner = C * 64;
    int bp = t / inner, rem = t % inner;
    int b = bp / NTOK, p = bp % NTOK;
    int c = rem >> 6, ij = rem & 63, i = ij >> 3, j = ij & 7;
    int ph = p / 15, pw = p % 15;
    out[t] = img[(((size_t)(b * C + c) * 120) + ph * 8 + i) * 120 + pw * 8 + j];
}

// ---------------------------------------------------------------------------
// Host orchestration
// ---------------------------------------------------------------------------
struct Lin { const float *w, *b; };
struct LNp { const float *g, *b; };
struct AttnP { LNp ln; const float* qkv; Lin out; };
struct FfnP { LNp ln; Lin fc1, fc2; };
struct CrossP { LNp ln; const float *q, *k, *v; Lin out; };
struct VitLayer { AttnP attn; FfnP ffn; };
struct CrossLayer { AttnP attn; CrossP cross; FfnP ffn; };
struct GapP { LNp ln; Lin fc1, fc2; };

extern "C" void kernel_launch(void* const* d_in, const int* in_sizes, int n_in,
                              void* d_out, int out_size, void* d_ws, size_t ws_size,
                              hipStream_t stream) {
    (void)in_sizes; (void)n_in; (void)out_size; (void)ws_size;
    const int Bn = 8, M = Bn * NTOK;                 // 1800 rows
    const float scale = 1.0f / sqrtf((float)HDIM);

    // ---- parse inputs (setup_inputs dict order, depth-first) ----
    const float* SAR = (const float*)d_in[0];
    const float* OPT = (const float*)d_in[1];
    int cur = 2;
    auto nxt = [&]() { return (const float*)d_in[cur++]; };
    auto pLN = [&](LNp& l) { l.g = nxt(); l.b = nxt(); };
    auto pLin = [&](Lin& l) { l.w = nxt(); l.b = nxt(); };
    auto pAttn = [&](AttnP& a) { pLN(a.ln); a.qkv = nxt(); pLin(a.out); };
    auto pFfn = [&](FfnP& f) { pLN(f.ln); pLin(f.fc1); pLin(f.fc2); };

    Lin s1_inp; VitLayer s1L[6]; LNp s1_out;
    pLin(s1_inp);
    for (int i = 0; i < 6; ++i) { pAttn(s1L[i].attn); pFfn(s1L[i].ffn); }
    pLN(s1_out);

    Lin s2_inp; VitLayer s2L[12]; LNp s2_out;
    pLin(s2_inp);
    for (int i = 0; i < 12; ++i) { pAttn(s2L[i].attn); pFfn(s2L[i].ffn); }
    pLN(s2_out);

    CrossLayer cL[6]; LNp c_out;
    for (int i = 0; i < 6; ++i) {
        pAttn(cL[i].attn);
        pLN(cL[i].cross.ln);
        cL[i].cross.q = nxt(); cL[i].cross.k = nxt(); cL[i].cross.v = nxt();
        pLin(cL[i].cross.out);
        pFfn(cL[i].ffn);
    }
    pLN(c_out);

    GapP gap1; pLN(gap1.ln); pLin(gap1.fc1); pLin(gap1.fc2);
    GapP gap2; pLN(gap2.ln); pLin(gap2.fc1); pLin(gap2.fc2);
    const float* abias = nxt();                      // [1,16,225,225]

    // ---- workspace ----
    float* w = (float*)d_ws; size_t off = 0;
    auto alloc = [&](size_t n) { float* p = w + off; off += n; return p; };
    float* xA      = alloc((size_t)M * DIMM);
    float* xn      = alloc((size_t)M * DIMM);
    float* cn      = alloc((size_t)M * DIMM);
    float* qkvb    = alloc((size_t)M * 2304);
    float* scores  = alloc((size_t)Bn * NHEAD * NTOK * SLD);
    float* attnout = alloc((size_t)M * DIMM);
    float* hbuf    = alloc((size_t)M * 3072);
    float* patches = hbuf;                           // alias: disjoint in time
    float* pooled  = alloc((size_t)Bn * DIMM);
    float* pooledn = alloc((size_t)Bn * DIMM);
    float* gh      = alloc((size_t)Bn * 3072);

    // ---- outputs (tuple order, flat) ----
    float* o = (float*)d_out;
    float* sar_enc   = o;                 o += (size_t)M * DIMM;
    float* sar_gap   = o;                 o += (size_t)Bn * DIMM;
    float* opt_enc   = o;                 o += (size_t)M * DIMM;
    float* opt_gap   = o;                 o += (size_t)Bn * DIMM;
    float* joint_enc = o;                 o += (size_t)M * DIMM;
    float* joint_gap = o;

    // ---- launch helpers ----
    auto gemm = [&](const float* A, int lda, const float* W, int ldw,
                    const float* bias, const float* res, int ldres,
                    float* C, int ldc, int Mm, int Nn, int Kk, int flags) {
        dim3 g((Nn + 127) / 128, (Mm + 63) / 64);
        gemm_wmma<<<g, 256, 0, stream>>>(A, lda, W, ldw, bias, res, ldres,
                                         C, ldc, Mm, Nn, Kk, flags);
    };
    auto layernorm = [&](const float* X, const LNp& l, float* Y, int rows) {
        layernorm_rows<<<(rows + 7) / 8, 256, 0, stream>>>(X, l.g, l.b, Y, rows);
    };
    auto attention = [&](float* xio, const Lin& proj) {
        attn_scores_wmma<<<dim3(2, 4, Bn * NHEAD), 256, 0, stream>>>(qkvb, abias, scores, scale);
        softmax_rows<<<(Bn * NHEAD * NTOK + 7) / 8, 256, 0, stream>>>(scores, Bn * NHEAD * NTOK);
        attn_av_wmma<<<dim3(1, 4, Bn * NHEAD), 128, 0, stream>>>(scores, qkvb, attnout);
        gemm(attnout, DIMM, proj.w, DIMM, proj.b, xio, DIMM, xio, DIMM, M, DIMM, DIMM, 1 | 4);
    };
    auto self_attn = [&](float* x, const AttnP& p) {
        layernorm(x, p.ln, xn, M);
        gemm(xn, DIMM, p.qkv, 2304, nullptr, nullptr, 0, qkvb, 2304, M, 2304, DIMM, 0);
        attention(x, p.out);
    };
    auto ffn = [&](float* x, const FfnP& p) {
        layernorm(x, p.ln, xn, M);
        gemm(xn, DIMM, p.fc1.w, 3072, p.fc1.b, nullptr, 0, hbuf, 3072, M, 3072, DIMM, 1 | 2);
        gemm(hbuf, 3072, p.fc2.w, DIMM, p.fc2.b, x, DIMM, x, DIMM, M, DIMM, 3072, 1 | 4);
    };
    auto vit = [&](const float* imgs, int C, const Lin& inp, const VitLayer* L,
                   int depth, const LNp& lnout, float* enc) {
        int total = Bn * NTOK * C * 64;
        patchify_k<<<(total + 255) / 256, 256, 0, stream>>>(imgs, patches, C, total);
        gemm(patches, C * 64, inp.w, DIMM, inp.b, nullptr, 0, xA, DIMM, M, DIMM, C * 64, 1);
        for (int i = 0; i < depth; ++i) { self_attn(xA, L[i].attn); ffn(xA, L[i].ffn); }
        layernorm(xA, lnout, enc, M);
    };
    auto gap_head = [&](const float* enc, const GapP& g, float* out_) {
        mean_pool_k<<<(Bn * DIMM + 255) / 256, 256, 0, stream>>>(enc, pooled, Bn);
        layernorm(pooled, g.ln, pooledn, Bn);
        gemm(pooledn, DIMM, g.fc1.w, 3072, g.fc1.b, nullptr, 0, gh, 3072, Bn, 3072, DIMM, 1 | 2);
        gemm(gh, 3072, g.fc2.w, DIMM, g.fc2.b, nullptr, 0, out_, DIMM, Bn, DIMM, 3072, 1);
    };

    // ---- pipeline ----
    vit(SAR, 2, s1_inp, s1L, 6, s1_out, sar_enc);
    gap_head(sar_enc, gap1, sar_gap);
    vit(OPT, 12, s2_inp, s2L, 12, s2_out, opt_enc);
    gap_head(opt_enc, gap2, opt_gap);

    hipMemcpyAsync(xA, sar_enc, (size_t)M * DIMM * sizeof(float),
                   hipMemcpyDeviceToDevice, stream);
    for (int i = 0; i < 6; ++i) {
        self_attn(xA, cL[i].attn);
        // cross attention: q from x, k/v from opt_enc (shared ln params)
        layernorm(xA, cL[i].cross.ln, xn, M);
        layernorm(opt_enc, cL[i].cross.ln, cn, M);
        gemm(xn, DIMM, cL[i].cross.q, DIMM, nullptr, nullptr, 0, qkvb,          2304, M, DIMM, DIMM, 0);
        gemm(cn, DIMM, cL[i].cross.k, DIMM, nullptr, nullptr, 0, qkvb + DIMM,   2304, M, DIMM, DIMM, 0);
        gemm(cn, DIMM, cL[i].cross.v, DIMM, nullptr, nullptr, 0, qkvb + 2*DIMM, 2304, M, DIMM, DIMM, 0);
        attention(xA, cL[i].cross.out);
        ffn(xA, cL[i].ffn);
    }
    layernorm(xA, c_out, joint_enc, M);
    mean_pool_k<<<(Bn * DIMM + 255) / 256, 256, 0, stream>>>(joint_enc, joint_gap, Bn);
}
